// MultiTimeframeLNN_53876069761127
// MI455X (gfx1250) — compile-verified, matrix operand order checked by reading, not run on hardware
//
#include <hip/hip_runtime.h>
#include <hip/hip_bf16.h>
#include <cstdint>

#define HDIM 512
#define BSZ  256

typedef __attribute__((ext_vector_type(16))) _Float16 v16h;
typedef __attribute__((ext_vector_type(8)))  float    v8f;
typedef __attribute__((ext_vector_type(4)))  int      v4i;
typedef __attribute__((address_space(1))) const v4i*  gv4i_p;

union FragU { v4i q[2]; v16h v; };

__device__ __forceinline__ v16h frag_from_lds(const int* p) {
  FragU u;
  u.q[0] = ((const v4i*)p)[0];
  u.q[1] = ((const v4i*)p)[1];
  return u.v;
}
// Force global address space -> global_load_b128 (not flat_load), so weight
// traffic only uses LOADcnt and does not couple with the LDS DScnt pipe.
__device__ __forceinline__ v16h frag_from_glb(const unsigned* p) {
  gv4i_p gp = (gv4i_p)(const void*)p;
  FragU u;
  u.q[0] = gp[0];
  u.q[1] = gp[1];
  return u.v;
}
__device__ __forceinline__ unsigned pack2h(float a, float b) {
  union { _Float16 h[2]; unsigned u; } x;
  x.h[0] = (_Float16)a; x.h[1] = (_Float16)b;
  return x.u;
}
// 16-bit WMMA operand lane layout (ISA 7.12.2): VGPR v holds k-pairs
// {0..7,16..23} for lanes 0-15 half, +8 for lanes 16-31 half.
__device__ __forceinline__ int frag_k(int lane, int v) {
  int kk = (v < 4) ? (v * 2) : (16 + (v - 4) * 2);
  return kk + ((lane >> 4) << 3);
}
__device__ __forceinline__ v8f vzero() {
  v8f z = {0.f, 0.f, 0.f, 0.f, 0.f, 0.f, 0.f, 0.f};
  return z;
}
__device__ __forceinline__ float sigmoidf_(float x) { return 1.f / (1.f + expf(-x)); }

#define WMMA_F16(a, b, c) __builtin_amdgcn_wmma_f32_16x16x32_f16(false, (a), false, (b), (short)0, (c), false, false)
// Locality 3 -> WGP-scope prefetch (pull into near caches); SYS-scope prefetch
// would only fill GL2, where the packed weights already reside.
#define PREFETCH_NEAR(p) __builtin_prefetch((const void*)(p), 0, 3)

// ---------------------------------------------------------------------------
// Weight pre-pack: row-major W (N x K, out = in @ W^T) -> WMMA B-operand lane
// layout [ntile][kblock][lane][8 dwords] in fp16.  Two source segments with an
// optional zero-pad between them (used to align the x|h split to a k-block).
// ---------------------------------------------------------------------------
__global__ __launch_bounds__(256) void pack_w_kernel(
    const float* __restrict__ s1, int K1, int ld1, int off1,
    const float* __restrict__ s2, int K2, int ld2, int off2,
    int P, int KB, int total, unsigned* __restrict__ dst)
{
  for (int idx = blockIdx.x * 256 + threadIdx.x; idx < total; idx += gridDim.x * 256) {
    int v = idx & 7;
    int lane = (idx >> 3) & 31;
    int blk = idx >> 8;        // nt*KB + kb
    int kb = blk % KB;
    int nt = blk / KB;
    int n = nt * 16 + (lane & 15);
    int k0 = kb * 32 + frag_k(lane, v);
    float vals[2];
#pragma unroll
    for (int p = 0; p < 2; ++p) {
      int k = k0 + p;
      float x = 0.f;
      if (k < K1) x = s1[(size_t)n * ld1 + off1 + k];
      else if (k >= K1 + P && k < K1 + P + K2) x = s2[(size_t)n * ld2 + off2 + (k - K1 - P)];
      vals[p] = x;
    }
    dst[idx] = pack2h(vals[0], vals[1]);
  }
}

// ---------------------------------------------------------------------------
// Encoder: one block = one (timeframe, 16-row batch tile); 8 waves; h in LDS.
// ---------------------------------------------------------------------------
struct EncTF {
  const float* seq;
  const float* bg0; const float* lng0; const float* lnb0; const float* tau0;
  const float* bg1; const float* lng1; const float* lnb1; const float* tau1;
  const unsigned* wg0; const unsigned* wc0;   // layer0: Kp=544 (17 kb)
  const unsigned* wg1; const unsigned* wc1;   // layer1: Kp=1024 (32 kb)
  int T; int pad;
};
struct EncArgs { EncTF tf[5]; float* enc_out; };

// One LTC layer step.  K loop is split at kbSplit so every v_wmma executes
// unconditionally with full EXEC (no saveexec predication).
__device__ void ltc_layer(const int* As, int KB, int kbSplit,
                          const unsigned* wgp, const unsigned* wcp,
                          float* hbuf,
                          const float* bgl, const float* itl,
                          const float* lngl, const float* lnbl,
                          float* rmean, float* rrstd,
                          int tid, int lane, int wave)
{
  v8f accg[4], accx[4], accr[4];
#pragma unroll
  for (int j = 0; j < 4; ++j) { accg[j] = vzero(); accx[j] = vzero(); accr[j] = vzero(); }

  // x-part k-blocks: gate += A*Wg, xlin += A*[Win|0]
  for (int kb = 0; kb < kbSplit; ++kb) {
    v16h a = frag_from_lds(As + kb * 256 + lane * 8);
#pragma unroll
    for (int j = 0; j < 4; ++j) {
      int nt = wave + j * 8;
      size_t boff = ((size_t)(nt * KB + kb) * 32 + lane) * 8;  // dwords
      v16h bfg = frag_from_glb(wgp + boff);
      v16h bfc = frag_from_glb(wcp + boff);
      PREFETCH_NEAR(wgp + boff + 256);   // next kblock, speculative
      PREFETCH_NEAR(wcp + boff + 256);
      accg[j] = WMMA_F16(a, bfg, accg[j]);
      accx[j] = WMMA_F16(a, bfc, accx[j]);
    }
  }
  // h-part k-blocks: gate += A*Wg, rec += A*[0|Wrec]
  for (int kb = kbSplit; kb < KB; ++kb) {
    v16h a = frag_from_lds(As + kb * 256 + lane * 8);
#pragma unroll
    for (int j = 0; j < 4; ++j) {
      int nt = wave + j * 8;
      size_t boff = ((size_t)(nt * KB + kb) * 32 + lane) * 8;
      v16h bfg = frag_from_glb(wgp + boff);
      v16h bfc = frag_from_glb(wcp + boff);
      PREFETCH_NEAR(wgp + boff + 256);
      PREFETCH_NEAR(wcp + boff + 256);
      accg[j] = WMMA_F16(a, bfg, accg[j]);
      accr[j] = WMMA_F16(a, bfc, accr[j]);
    }
  }

  // C/D layout: VGPR r -> m = r + 8*(lane>=16), n = ntile*16 + (lane&15)
  int hi = lane >> 4, nlo = lane & 15;
#pragma unroll
  for (int j = 0; j < 4; ++j) {
    int n = (wave + j * 8) * 16 + nlo;
#pragma unroll
    for (int r = 0; r < 8; ++r) {
      int m = r + 8 * hi;
      float hol = hbuf[m * HDIM + n];
      float gate = sigmoidf_(tanhf(accg[j][r] + bgl[n]));
      float hraw = tanhf(hol + (-hol * itl[n] + accx[j][r] + gate * accr[j][r]));
      hbuf[m * HDIM + n] = hraw;
    }
  }
  __syncthreads();
  if (tid < 16) {
    float s = 0.f, ss = 0.f;
    for (int n = 0; n < HDIM; ++n) { float x = hbuf[tid * HDIM + n]; s += x; ss += x * x; }
    float mu = s * (1.f / HDIM);
    float var = ss * (1.f / HDIM) - mu * mu;
    rmean[tid] = mu; rrstd[tid] = rsqrtf(var + 1e-5f);
  }
  __syncthreads();
  for (int i = tid; i < 16 * HDIM; i += 256) {
    int m = i >> 9, n = i & 511;
    hbuf[i] = (hbuf[i] - rmean[m]) * rrstd[m] * lngl[n] + lnbl[n];
  }
  __syncthreads();
}

__global__ __launch_bounds__(256) void encoder_kernel(EncArgs args)
{
  const int tid = threadIdx.x, lane = tid & 31, wave = tid >> 5;
  const int tfi = blockIdx.x >> 4;          // 5 tfs x 16 batch tiles
  const int b0 = (blockIdx.x & 15) * 16;
  const EncTF tf = args.tf[tfi];

  extern __shared__ char smem_raw[];
  float* h0 = (float*)smem_raw;             // 16x512 f32
  float* h1 = h0 + 16 * HDIM;               // 16x512 f32
  int* As = (int*)(h1 + 16 * HDIM);         // 32 kblocks x 256 dwords (A staging)
  float* prm = (float*)(As + 32 * 256);     // 8x512 cached params
  float* rmean = prm + 8 * HDIM;
  float* rrstd = rmean + 16;
  float *bg0 = prm, *bg1 = prm + HDIM, *lng0 = prm + 2 * HDIM, *lng1 = prm + 3 * HDIM,
        *lnb0 = prm + 4 * HDIM, *lnb1 = prm + 5 * HDIM, *it0 = prm + 6 * HDIM, *it1 = prm + 7 * HDIM;

  for (int i = tid; i < HDIM; i += 256) {
    bg0[i] = tf.bg0[i]; bg1[i] = tf.bg1[i];
    lng0[i] = tf.lng0[i]; lng1[i] = tf.lng1[i];
    lnb0[i] = tf.lnb0[i]; lnb1[i] = tf.lnb1[i];
    it0[i] = 1.f / (log1pf(expf(tf.tau0[i])) + 1.f);   // 1/(softplus(tau)+1)
    it1[i] = 1.f / (log1pf(expf(tf.tau1[i])) + 1.f);
  }
  for (int i = tid; i < 16 * HDIM; i += 256) { h0[i] = 0.f; h1[i] = 0.f; }
  __syncthreads();

  const int T = tf.T;
#pragma unroll 1
  for (int t = 0; t < T; ++t) {
    // pack layer0 A = [x(16) | zeros(16) | h0(512)] -> Kp=544, 17 kblocks
    for (int idx = tid; idx < 17 * 256; idx += 256) {
      int v = idx & 7, ln = (idx >> 3) & 31, kb = idx >> 8;
      int m = ln & 15;
      int k0 = kb * 32 + frag_k(ln, v);
      const float* xr = tf.seq + ((size_t)(b0 + m) * T + t) * 16;
      float a0 = (k0 < 16) ? xr[k0] : (k0 >= 32 ? h0[m * HDIM + k0 - 32] : 0.f);
      int k1 = k0 + 1;
      float a1 = (k1 < 16) ? xr[k1] : (k1 >= 32 ? h0[m * HDIM + k1 - 32] : 0.f);
      As[idx] = pack2h(a0, a1);
    }
    __syncthreads();
    ltc_layer(As, 17, 1, tf.wg0, tf.wc0, h0, bg0, it0, lng0, lnb0, rmean, rrstd, tid, lane, wave);

    // pack layer1 A = [h0(512) | h1(512)] -> 32 kblocks
    for (int idx = tid; idx < 32 * 256; idx += 256) {
      int v = idx & 7, ln = (idx >> 3) & 31, kb = idx >> 8;
      int m = ln & 15;
      int k0 = kb * 32 + frag_k(ln, v);
      float a0 = (k0 < HDIM) ? h0[m * HDIM + k0] : h1[m * HDIM + k0 - HDIM];
      int k1 = k0 + 1;
      float a1 = (k1 < HDIM) ? h0[m * HDIM + k1] : h1[m * HDIM + k1 - HDIM];
      As[idx] = pack2h(a0, a1);
    }
    __syncthreads();
    ltc_layer(As, 32, 16, tf.wg1, tf.wc1, h1, bg1, it1, lng1, lnb1, rmean, rrstd, tid, lane, wave);
  }

  for (int i = tid; i < 16 * HDIM; i += 256) {
    int m = i >> 9, n = i & 511;
    args.enc_out[((size_t)tfi * BSZ + b0 + m) * HDIM + n] = h1[i];
  }
}

// ---------------------------------------------------------------------------
// Cross-timeframe attention: one block = 4 batch elements (20 rows, 2 M-tiles).
// QKV + O projections via WMMA; 5-token softmax attention in VALU.
// ---------------------------------------------------------------------------
__global__ __launch_bounds__(256) void attn_kernel(
    const float* __restrict__ enc_out,            // [5][256][512]
    const unsigned* __restrict__ pwq, const unsigned* __restrict__ pwk,
    const unsigned* __restrict__ pwv, const unsigned* __restrict__ pwo,
    const float* __restrict__ bo, const float* __restrict__ alng,
    const float* __restrict__ alnb, float* __restrict__ xattn)
{
  const int tid = threadIdx.x, lane = tid & 31, wave = tid >> 5;
  const int b0 = blockIdx.x * 4;
  extern __shared__ char smem_raw[];
  int* As = (int*)smem_raw;                       // 2 mt x 16 kb x 256
  _Float16* qh = (_Float16*)(As + 2 * 16 * 256);  // 20x512 each
  _Float16* kh = qh + 20 * 512;
  _Float16* vh = kh + 20 * 512;
  _Float16* ch = vh + 20 * 512;
  float* xa = (float*)(ch + 20 * 512);            // 20x512 f32 (pre-LN)
  float* rmean = xa + 20 * 512;
  float* rrstd = rmean + 20;
  const int hi = lane >> 4, nlo = lane & 15;

  // pack x rows (b,t) -> A
  for (int idx = tid; idx < 2 * 16 * 256; idx += 256) {
    int v = idx & 7, ln = (idx >> 3) & 31, kb = (idx >> 8) & 15, mt = idx >> 12;
    int row = mt * 16 + (ln & 15);
    int k0 = kb * 32 + frag_k(ln, v);
    float a0 = 0.f, a1 = 0.f;
    if (row < 20) {
      int b = b0 + row / 5, t = row % 5;
      const float* xr = enc_out + ((size_t)t * BSZ + b) * HDIM;
      a0 = xr[k0]; a1 = xr[k0 + 1];
    }
    As[idx] = pack2h(a0, a1);
  }
  __syncthreads();

  for (int mt = 0; mt < 2; ++mt) {
    v8f aq[4], ak[4], av[4];
#pragma unroll
    for (int j = 0; j < 4; ++j) { aq[j] = vzero(); ak[j] = vzero(); av[j] = vzero(); }
    for (int kb = 0; kb < 16; ++kb) {
      v16h a = frag_from_lds(As + (mt * 16 + kb) * 256 + lane * 8);
#pragma unroll
      for (int j = 0; j < 4; ++j) {
        int nt = wave + j * 8;
        size_t boff = ((size_t)(nt * 16 + kb) * 32 + lane) * 8;
        aq[j] = WMMA_F16(a, frag_from_glb(pwq + boff), aq[j]);
        ak[j] = WMMA_F16(a, frag_from_glb(pwk + boff), ak[j]);
        av[j] = WMMA_F16(a, frag_from_glb(pwv + boff), av[j]);
      }
    }
#pragma unroll
    for (int j = 0; j < 4; ++j) {
      int n = (wave + j * 8) * 16 + nlo;
#pragma unroll
      for (int r = 0; r < 8; ++r) {
        int row = mt * 16 + r + 8 * hi;
        if (row < 20) {
          qh[row * 512 + n] = (_Float16)aq[j][r];
          kh[row * 512 + n] = (_Float16)ak[j][r];
          vh[row * 512 + n] = (_Float16)av[j][r];
        }
      }
    }
  }
  __syncthreads();

  // attention: thread = (b_local, head, q)
  if (tid < 160) {
    int b = tid / 40, rem = tid % 40, hh = rem / 5, q = rem % 5;
    int rq = b * 5 + q, off = hh * 64;
    float s[5], mx = -1e30f;
    for (int kk = 0; kk < 5; ++kk) {
      float acc = 0.f;
      for (int d = 0; d < 64; ++d)
        acc += (float)qh[rq * 512 + off + d] * (float)kh[(b * 5 + kk) * 512 + off + d];
      s[kk] = acc * 0.125f;                      // 1/sqrt(64)
      mx = fmaxf(mx, s[kk]);
    }
    float den = 0.f;
    for (int kk = 0; kk < 5; ++kk) { s[kk] = expf(s[kk] - mx); den += s[kk]; }
    float inv = 1.f / den;
    for (int d = 0; d < 64; ++d) {
      float c = 0.f;
      for (int kk = 0; kk < 5; ++kk) c += s[kk] * (float)vh[(b * 5 + kk) * 512 + off + d];
      ch[rq * 512 + off + d] = (_Float16)(c * inv);
    }
  }
  __syncthreads();

  // pack ctx -> A, then O projection
  for (int idx = tid; idx < 2 * 16 * 256; idx += 256) {
    int v = idx & 7, ln = (idx >> 3) & 31, kb = (idx >> 8) & 15, mt = idx >> 12;
    int row = mt * 16 + (ln & 15);
    int k0 = kb * 32 + frag_k(ln, v);
    float a0 = 0.f, a1 = 0.f;
    if (row < 20) { a0 = (float)ch[row * 512 + k0]; a1 = (float)ch[row * 512 + k0 + 1]; }
    As[idx] = pack2h(a0, a1);
  }
  __syncthreads();
  for (int mt = 0; mt < 2; ++mt) {
    v8f ao[4];
#pragma unroll
    for (int j = 0; j < 4; ++j) ao[j] = vzero();
    for (int kb = 0; kb < 16; ++kb) {
      v16h a = frag_from_lds(As + (mt * 16 + kb) * 256 + lane * 8);
#pragma unroll
      for (int j = 0; j < 4; ++j) {
        int nt = wave + j * 8;
        size_t boff = ((size_t)(nt * 16 + kb) * 32 + lane) * 8;
        ao[j] = WMMA_F16(a, frag_from_glb(pwo + boff), ao[j]);
      }
    }
#pragma unroll
    for (int j = 0; j < 4; ++j) {
      int n = (wave + j * 8) * 16 + nlo;
#pragma unroll
      for (int r = 0; r < 8; ++r) {
        int row = mt * 16 + r + 8 * hi;
        if (row < 20) {
          int b = b0 + row / 5, t = row % 5;
          xa[row * 512 + n] = enc_out[((size_t)t * BSZ + b) * HDIM + n] + ao[j][r] + bo[n];
        }
      }
    }
  }
  __syncthreads();
  if (tid < 20) {
    float s = 0.f, ss = 0.f;
    for (int n = 0; n < HDIM; ++n) { float x = xa[tid * HDIM + n]; s += x; ss += x * x; }
    float mu = s * (1.f / HDIM), var = ss * (1.f / HDIM) - mu * mu;
    rmean[tid] = mu; rrstd[tid] = rsqrtf(var + 1e-5f);
  }
  __syncthreads();
  for (int i = tid; i < 20 * HDIM; i += 256) {
    int row = i >> 9, n = i & 511;
    int b = b0 + row / 5, t = row % 5;
    xattn[((size_t)b * 5 + t) * HDIM + n] = (xa[i] - rmean[row]) * rrstd[row] * alng[n] + alnb[n];
  }
}

// ---------------------------------------------------------------------------
// Fusion + classifier: one block = 16 batch rows. K=2592 = 81*32 exactly.
// ---------------------------------------------------------------------------
__global__ __launch_bounds__(256) void head_kernel(
    const float* __restrict__ xattn, const float* __restrict__ ctxf,
    const unsigned* __restrict__ pfw, const float* __restrict__ fb,
    const unsigned* __restrict__ pw1, const float* __restrict__ cb1,
    const float* __restrict__ w2, const float* __restrict__ cb2,
    float* __restrict__ out)
{
  const int tid = threadIdx.x, lane = tid & 31, wave = tid >> 5;
  const int b0 = blockIdx.x * 16;
  extern __shared__ char smem_raw[];
  int* As = (int*)smem_raw;                       // 81 kb x 256
  _Float16* fh = (_Float16*)(As + 81 * 256);      // 16x512
  float* hm = (float*)(fh + 16 * 512);            // 16x256
  const int hi = lane >> 4, nlo = lane & 15;

  for (int idx = tid; idx < 81 * 256; idx += 256) {
    int v = idx & 7, ln = (idx >> 3) & 31, kb = idx >> 8;
    int m = ln & 15, b = b0 + m;
    int k0 = kb * 32 + frag_k(ln, v);
    float a0 = (k0 < 2560) ? xattn[(size_t)b * 2560 + k0]
             : ((k0 < 2592) ? ctxf[b * 32 + k0 - 2560] : 0.f);
    int k1 = k0 + 1;
    float a1 = (k1 < 2560) ? xattn[(size_t)b * 2560 + k1]
             : ((k1 < 2592) ? ctxf[b * 32 + k1 - 2560] : 0.f);
    As[idx] = pack2h(a0, a1);
  }
  __syncthreads();

  {
    v8f acc[4];
#pragma unroll
    for (int j = 0; j < 4; ++j) acc[j] = vzero();
    for (int kb = 0; kb < 81; ++kb) {
      v16h a = frag_from_lds(As + kb * 256 + lane * 8);
#pragma unroll
      for (int j = 0; j < 4; ++j) {
        int nt = wave + j * 8;
        size_t boff = ((size_t)(nt * 81 + kb) * 32 + lane) * 8;
        acc[j] = WMMA_F16(a, frag_from_glb(pfw + boff), acc[j]);
      }
    }
#pragma unroll
    for (int j = 0; j < 4; ++j) {
      int n = (wave + j * 8) * 16 + nlo;
#pragma unroll
      for (int r = 0; r < 8; ++r) {
        int m = r + 8 * hi;
        fh[m * 512 + n] = (_Float16)fmaxf(acc[j][r] + fb[n], 0.f);
      }
    }
  }
  __syncthreads();

  for (int idx = tid; idx < 16 * 256; idx += 256) {
    int v = idx & 7, ln = (idx >> 3) & 31, kb = idx >> 8;
    int m = ln & 15;
    int k0 = kb * 32 + frag_k(ln, v);
    As[idx] = pack2h((float)fh[m * 512 + k0], (float)fh[m * 512 + k0 + 1]);
  }
  __syncthreads();

  {
    v8f acc[2]; acc[0] = vzero(); acc[1] = vzero();
    for (int kb = 0; kb < 16; ++kb) {
      v16h a = frag_from_lds(As + kb * 256 + lane * 8);
#pragma unroll
      for (int j = 0; j < 2; ++j) {
        int nt = wave + j * 8;
        size_t boff = ((size_t)(nt * 16 + kb) * 32 + lane) * 8;
        acc[j] = WMMA_F16(a, frag_from_glb(pw1 + boff), acc[j]);
      }
    }
#pragma unroll
    for (int j = 0; j < 2; ++j) {
      int n = (wave + j * 8) * 16 + nlo;
#pragma unroll
      for (int r = 0; r < 8; ++r) {
        int m = r + 8 * hi;
        hm[m * 256 + n] = fmaxf(acc[j][r] + cb1[n], 0.f);
      }
    }
  }
  __syncthreads();
  if (tid < 16) {
    float acc = cb2[0];
    for (int n = 0; n < 256; ++n) acc += hm[tid * 256 + n] * w2[n];
    out[b0 + tid] = sigmoidf_(acc);
  }
}

// ---------------------------------------------------------------------------
// Host launch.  Param indices assume jax pytree (sorted-key) flatten order:
// seqs(0-4), context(5), attn{Wk,Wo,Wq,Wv,bo,ln_b,ln_g}(6-12),
// cls{W1,W2,b1,b2}(13-16), enc{15min,1day,4hour,5min,60min} x 2 layers x
// {Wg,Win,Wrec,bg,ln_b,ln_g,tau}(17-86), fusion{W,b}(87-88).
// ---------------------------------------------------------------------------
extern "C" void kernel_launch(void* const* d_in, const int* in_sizes, int n_in,
                              void* d_out, int out_size, void* d_ws, size_t ws_size,
                              hipStream_t stream)
{
  const float* seq[5] = { (const float*)d_in[0], (const float*)d_in[1],
                          (const float*)d_in[2], (const float*)d_in[3],
                          (const float*)d_in[4] };
  const int Ts[5] = {288, 96, 48, 36, 30};
  const float* ctxf = (const float*)d_in[5];
  const float* aWk = (const float*)d_in[6];
  const float* aWo = (const float*)d_in[7];
  const float* aWq = (const float*)d_in[8];
  const float* aWv = (const float*)d_in[9];
  const float* abo = (const float*)d_in[10];
  const float* alnb = (const float*)d_in[11];
  const float* alng = (const float*)d_in[12];
  const float* cW1 = (const float*)d_in[13];
  const float* cW2 = (const float*)d_in[14];
  const float* cb1 = (const float*)d_in[15];
  const float* cb2 = (const float*)d_in[16];
  const int encBase[5] = {59, 17, 73, 45, 31};  // TFS order -> sorted-name base
  const float* fW = (const float*)d_in[87];
  const float* fb = (const float*)d_in[88];

  char* ws = (char*)d_ws;
  size_t off = 0;
  auto alloc = [&](size_t bytes) -> void* {
    void* p = ws + off;
    off += (bytes + 255) & ~(size_t)255;
    return p;
  };

  unsigned *wg0[5], *wc0[5], *wg1[5], *wc1[5];
  for (int i = 0; i < 5; ++i) {
    wg0[i] = (unsigned*)alloc((size_t)32 * 17 * 256 * 4);
    wc0[i] = (unsigned*)alloc((size_t)32 * 17 * 256 * 4);
    wg1[i] = (unsigned*)alloc((size_t)32 * 32 * 256 * 4);
    wc1[i] = (unsigned*)alloc((size_t)32 * 32 * 256 * 4);
  }
  unsigned* pwq = (unsigned*)alloc((size_t)32 * 16 * 256 * 4);
  unsigned* pwk = (unsigned*)alloc((size_t)32 * 16 * 256 * 4);
  unsigned* pwv = (unsigned*)alloc((size_t)32 * 16 * 256 * 4);
  unsigned* pwo = (unsigned*)alloc((size_t)32 * 16 * 256 * 4);
  unsigned* pfw = (unsigned*)alloc((size_t)32 * 81 * 256 * 4);
  unsigned* pw1 = (unsigned*)alloc((size_t)16 * 16 * 256 * 4);
  float* enc_out = (float*)alloc((size_t)5 * 256 * 512 * 4);
  float* xattn = (float*)alloc((size_t)256 * 5 * 512 * 4);

  auto packLaunch = [&](const float* s1, int K1, int ld1, int o1,
                        const float* s2, int K2, int ld2, int o2,
                        int P, int N, int KB, unsigned* dst) {
    int total = (N / 16) * KB * 256;
    pack_w_kernel<<<(N / 16) * KB, 256, 0, stream>>>(s1, K1, ld1, o1, s2, K2, ld2, o2, P, KB, total, dst);
  };

  for (int i = 0; i < 5; ++i) {
    int pb = encBase[i];
    const float* Wg0 = (const float*)d_in[pb + 0];
    const float* Wi0 = (const float*)d_in[pb + 1];
    const float* Wr0 = (const float*)d_in[pb + 2];
    const float* Wg1 = (const float*)d_in[pb + 7];
    const float* Wi1 = (const float*)d_in[pb + 8];
    const float* Wr1 = (const float*)d_in[pb + 9];
    // layer0: [x(16) | pad(16) | h(512)] -> Kp=544
    packLaunch(Wg0, 16, 528, 0, Wg0, 512, 528, 16, 16, 512, 17, wg0[i]);
    packLaunch(Wi0, 16, 16, 0, Wr0, 512, 512, 0, 16, 512, 17, wc0[i]);
    // layer1: [h_prev(512) | h(512)] -> Kp=1024
    packLaunch(Wg1, 1024, 1024, 0, Wg1, 0, 1024, 0, 0, 512, 32, wg1[i]);
    packLaunch(Wi1, 512, 512, 0, Wr1, 512, 512, 0, 0, 512, 32, wc1[i]);
  }
  packLaunch(aWq, 512, 512, 0, aWq, 0, 512, 0, 0, 512, 16, pwq);
  packLaunch(aWk, 512, 512, 0, aWk, 0, 512, 0, 0, 512, 16, pwk);
  packLaunch(aWv, 512, 512, 0, aWv, 0, 512, 0, 0, 512, 16, pwv);
  packLaunch(aWo, 512, 512, 0, aWo, 0, 512, 0, 0, 512, 16, pwo);
  packLaunch(fW, 2592, 2592, 0, fW, 0, 2592, 0, 0, 512, 81, pfw);
  packLaunch(cW1, 512, 512, 0, cW1, 0, 512, 0, 0, 256, 16, pw1);

  EncArgs ea;
  for (int i = 0; i < 5; ++i) {
    int pb = encBase[i];
    EncTF& e = ea.tf[i];
    e.seq = seq[i]; e.T = Ts[i]; e.pad = 0;
    e.bg0 = (const float*)d_in[pb + 3];  e.lnb0 = (const float*)d_in[pb + 4];
    e.lng0 = (const float*)d_in[pb + 5]; e.tau0 = (const float*)d_in[pb + 6];
    e.bg1 = (const float*)d_in[pb + 10]; e.lnb1 = (const float*)d_in[pb + 11];
    e.lng1 = (const float*)d_in[pb + 12]; e.tau1 = (const float*)d_in[pb + 13];
    e.wg0 = wg0[i]; e.wc0 = wc0[i]; e.wg1 = wg1[i]; e.wc1 = wc1[i];
  }
  ea.enc_out = enc_out;

  size_t smem_enc = (size_t)(16 * 512 * 2) * 4 + 32 * 256 * 4 + 8 * 512 * 4 + 64 * 4;
  hipFuncSetAttribute((const void*)encoder_kernel, hipFuncAttributeMaxDynamicSharedMemorySize, (int)smem_enc);
  encoder_kernel<<<80, 256, smem_enc, stream>>>(ea);

  size_t smem_attn = (size_t)2 * 16 * 256 * 4 + 4 * (20 * 512 * 2) + 20 * 512 * 4 + 64 * 4;
  hipFuncSetAttribute((const void*)attn_kernel, hipFuncAttributeMaxDynamicSharedMemorySize, (int)smem_attn);
  attn_kernel<<<64, 256, smem_attn, stream>>>(enc_out, pwq, pwk, pwv, pwo, abo, alng, alnb, xattn);

  size_t smem_head = (size_t)81 * 256 * 4 + 16 * 512 * 2 + 16 * 256 * 4;
  hipFuncSetAttribute((const void*)head_kernel, hipFuncAttributeMaxDynamicSharedMemorySize, (int)smem_head);
  head_kernel<<<16, 256, smem_head, stream>>>(xattn, ctxf, pfw, fb, pw1, cb1, cW2, cb2, (float*)d_out);

  (void)in_sizes; (void)n_in; (void)out_size; (void)ws_size;
}
